// RN4_54820962566529
// MI455X (gfx1250) — matrix operand order repaired and checked by reading, not compiled, not run on hardware
//
#include <hip/hip_runtime.h>
#include <hip/hip_bf16.h>

// ---------------------------------------------------------------------------
// Types for WMMA fragments (wave32, gfx1250)
// ---------------------------------------------------------------------------
typedef __attribute__((ext_vector_type(16))) __bf16 v16bf;
typedef __attribute__((ext_vector_type(8)))  float  v8f;

union BFrag {
    v16bf  v;
    float4 f[2];   // 2 x 16B chunks = 32B = 16 bf16 halves across 8 VGPRs
};

// ---------------------------------------------------------------------------
// CDNA5 async global->LDS copy (ASYNCcnt path), per CDNA5 ISA
// and CDNA5 ISA (GLOBAL_LOAD_ASYNC_TO_LDS_B128, opcode 98).
// VDST VGPR = LDS byte address (low 32 bits of a generic LDS pointer),
// VADDR = 64-bit global address. Completion via s_wait_asynccnt.
// ---------------------------------------------------------------------------
__device__ __forceinline__ void async_load_b128(unsigned lds_byte_addr, const void* gaddr) {
    asm volatile("global_load_async_to_lds_b128 %0, %1, off"
                 :: "v"(lds_byte_addr), "v"((unsigned long long)(uintptr_t)gaddr)
                 : "memory");
}
__device__ __forceinline__ void wait_asynccnt0() {
    asm volatile("s_wait_asynccnt 0x0" ::: "memory");
}

// ---------------------------------------------------------------------------
// Fragment loaders, per CDNA5 ISA 7.12.2 layouts (wave32)
//
// A (16x32 bf16, MxK): lane l -> M = mbase + (l&15), hi = l>>4.
//   per lane: 8 contiguous halves at k0+hi*8, then 8 at k0+16+hi*8.
//
// B (32x16 bf16, KxN): lane l -> N = nbase + (l&15), hi = l>>4.
//   per lane: 8 contiguous halves at k0+hi*8, then 8 at k0+16+hi*8.
// ---------------------------------------------------------------------------
__device__ __forceinline__ v16bf load_a_frag(const __bf16* __restrict__ A,
                                             int lda, int mrow, int k0, int hi) {
    const __bf16* p = A + (size_t)mrow * lda + k0 + hi * 8;
    BFrag f;
    f.f[0] = *reinterpret_cast<const float4*>(p);
    f.f[1] = *reinterpret_cast<const float4*>(p + 16);
    return f.v;
}

// B fragment out of the LDS-staged tile. Row pitch padded to 40 halves (80B)
// to break the power-of-two bank stride of the 64B-per-row layout.
constexpr int B_ROW = 40;
__device__ __forceinline__ v16bf load_b_frag_lds(const __bf16* s, int n_local, int hi) {
    const __bf16* p = s + n_local * B_ROW + hi * 8;
    BFrag f;
    f.f[0] = *reinterpret_cast<const float4*>(p);
    f.f[1] = *reinterpret_cast<const float4*>(p + 16);
    return f.v;
}

__device__ __forceinline__ v8f wmma_bf16(const v16bf& a, const v16bf& b, const v8f& c) {
    return __builtin_amdgcn_wmma_f32_16x16x32_bf16(false, a, false, b,
                                                   (short)0, c, false, false);
}

// ---------------------------------------------------------------------------
// Split-bf16 GEMM:  out[m,n] = epilogue( sum_k A[m,k] * W[n,k] + bias[n] )
// A given as (Ah + Al), W as (Wh + Wl); product approximated as
//   Ah*Wh + Ah*Wl + Al*Wh   (f32 accumulate; drops only lo*lo ~ 2^-16 rel)
//
// Block: 256 threads = 8 waves, tile 128(M) x 64(N), K-step 32.
// B tile (h+l, 64 rows x 32 k) staged in LDS via async global->LDS copies,
// double buffered; every thread moves one 16B chunk per plane per step.
// Each wave owns a 16x64 strip: 4 acc tiles, 12 WMMAs per K-step.
//
// The compute guard for partial-M launches is forced through readfirstlane
// so it lowers as a SCALAR branch (s_cbranch), never EXEC masking: WMMA
// requires EXEC all-1s and is NOT skipped when EXEC==0 (ISA 4.2.2 / 7.12).
// Inactive waves still issue their share of async loads and hit every
// barrier, keeping __syncthreads valid block-wide.
// ---------------------------------------------------------------------------
template<bool RELU, bool SPLIT_OUT>
__global__ __launch_bounds__(256)
void gemm_split_bf16(const __bf16* __restrict__ Ah, const __bf16* __restrict__ Al, int lda,
                     const __bf16* __restrict__ Wh, const __bf16* __restrict__ Wl, int ldw,
                     const float*  __restrict__ bias,
                     float*  __restrict__ outF,
                     __bf16* __restrict__ outH, __bf16* __restrict__ outL,
                     int M, int N, int K, int ldo)
{
    __shared__ __align__(16) __bf16 sB[2][2][64 * B_ROW];   // [buf][h/l][n*40+k] = 20KB

    const int tid   = threadIdx.x;
    const int lane  = tid & 31;
    const int wave  = tid >> 5;
    const int hi    = lane >> 4;
    const int l15   = lane & 15;
    const int mbase = blockIdx.y * 128 + wave * 16;
    const int nbase = blockIdx.x * 64;
    // Wave-uniform guard as an SGPR value -> scalar branch, not EXEC mask.
    const bool mvalid = (__builtin_amdgcn_readfirstlane(mbase) < M);
    const int mrow  = mbase + l15;

    // --- async B-stage mapping: thread t moves 16B chunk (n_loc, kq..kq+7) ---
    const int n_loc = tid >> 2;             // 0..63
    const int kq    = (tid & 3) * 8;        // 0,8,16,24 (halves)
    const __bf16* gBh = Wh + (size_t)(nbase + n_loc) * ldw + kq;
    const __bf16* gBl = Wl + (size_t)(nbase + n_loc) * ldw + kq;
    const unsigned ldsOffChunk = (unsigned)(n_loc * B_ROW + kq) * 2u;  // bytes
    unsigned ldsH[2], ldsL[2];
    ldsH[0] = (unsigned)(uintptr_t)&sB[0][0][0] + ldsOffChunk;
    ldsL[0] = (unsigned)(uintptr_t)&sB[0][1][0] + ldsOffChunk;
    ldsH[1] = (unsigned)(uintptr_t)&sB[1][0][0] + ldsOffChunk;
    ldsL[1] = (unsigned)(uintptr_t)&sB[1][1][0] + ldsOffChunk;

    auto issue = [&](int buf, int k0) {
        async_load_b128(ldsH[buf], gBh + k0);
        async_load_b128(ldsL[buf], gBl + k0);
    };

    v8f acc0 = {}, acc1 = {}, acc2 = {}, acc3 = {};

    const int nk = K / 32;
    issue(0, 0);
    wait_asynccnt0();
    __syncthreads();

    for (int kt = 0; kt < nk; ++kt) {
        const int k0 = kt * 32;
        if (kt + 1 < nk) issue((kt + 1) & 1, k0 + 32);

        if (mvalid) {
            v16bf aH = load_a_frag(Ah, lda, mrow, k0, hi);
            v16bf aL = load_a_frag(Al, lda, mrow, k0, hi);
            if (kt + 1 < nk) {   // pull next A k-tile toward the near cache
                __builtin_prefetch(Ah + (size_t)mrow * lda + (k0 + 32), 0, 3);
                __builtin_prefetch(Al + (size_t)mrow * lda + (k0 + 32), 0, 3);
            }
            const __bf16* sbh = &sB[kt & 1][0][0];
            const __bf16* sbl = &sB[kt & 1][1][0];
            {
                v16bf bH = load_b_frag_lds(sbh, l15 +  0, hi);
                v16bf bL = load_b_frag_lds(sbl, l15 +  0, hi);
                acc0 = wmma_bf16(aH, bH, acc0);
                acc0 = wmma_bf16(aH, bL, acc0);
                acc0 = wmma_bf16(aL, bH, acc0);
            }
            {
                v16bf bH = load_b_frag_lds(sbh, l15 + 16, hi);
                v16bf bL = load_b_frag_lds(sbl, l15 + 16, hi);
                acc1 = wmma_bf16(aH, bH, acc1);
                acc1 = wmma_bf16(aH, bL, acc1);
                acc1 = wmma_bf16(aL, bH, acc1);
            }
            {
                v16bf bH = load_b_frag_lds(sbh, l15 + 32, hi);
                v16bf bL = load_b_frag_lds(sbl, l15 + 32, hi);
                acc2 = wmma_bf16(aH, bH, acc2);
                acc2 = wmma_bf16(aH, bL, acc2);
                acc2 = wmma_bf16(aL, bH, acc2);
            }
            {
                v16bf bH = load_b_frag_lds(sbh, l15 + 48, hi);
                v16bf bL = load_b_frag_lds(sbl, l15 + 48, hi);
                acc3 = wmma_bf16(aH, bH, acc3);
                acc3 = wmma_bf16(aH, bL, acc3);
                acc3 = wmma_bf16(aL, bH, acc3);
            }
        }

        if (kt + 1 < nk) {
            wait_asynccnt0();    // own async loads done (next buffer filled)
            __syncthreads();     // whole block done reading current buffer
        }
    }

    if (!mvalid) return;

    // C/D layout: VGPR r -> row mbase + r + hi*8, col = ntile + (lane&15)
    const int row0 = mbase + hi * 8;
    auto store_tile = [&](const v8f& a, int ntile) {
        const int col = ntile + l15;
        const float bv = bias ? bias[col] : 0.0f;
        #pragma unroll
        for (int r = 0; r < 8; ++r) {
            float xv = a[r] + bv;
            if (RELU) xv = fmaxf(xv, 0.0f);
            const size_t idx = (size_t)(row0 + r) * ldo + col;
            if (SPLIT_OUT) {
                __bf16 h = (__bf16)xv;
                outH[idx] = h;
                outL[idx] = (__bf16)(xv - (float)h);
            } else {
                outF[idx] = xv;
            }
        }
    };
    store_tile(acc0, nbase +  0);
    store_tile(acc1, nbase + 16);
    store_tile(acc2, nbase + 32);
    store_tile(acc3, nbase + 48);
}

// ---------------------------------------------------------------------------
// f32 -> (bf16 hi, bf16 lo) split
// ---------------------------------------------------------------------------
__global__ __launch_bounds__(256)
void split_bf16_kernel(const float* __restrict__ in,
                       __bf16* __restrict__ h, __bf16* __restrict__ l, size_t n)
{
    size_t i = (size_t)blockIdx.x * blockDim.x + threadIdx.x;
    if (i < n) {
        float x = in[i];
        __bf16 hh = (__bf16)x;
        h[i] = hh;
        l[i] = (__bf16)(x - (float)hh);
    }
}

// ---------------------------------------------------------------------------
// x[b,k1,v] = sum_k2 relu(a[b,k1,v] + bpart[b,k2,v] + c[b,v])
// bpart column tile staged in LDS (36x256 f32 = 36 KB) -> 36x reuse over k2.
// Output written as split bf16 for the final WMMA GEMM.
// ---------------------------------------------------------------------------
__global__ __launch_bounds__(256)
void pairsum_kernel(const float* __restrict__ a, const float* __restrict__ bp,
                    const float* __restrict__ c,
                    __bf16* __restrict__ xH, __bf16* __restrict__ xL)
{
    constexpr int Kk = 36, V = 2048;
    const int b   = blockIdx.y;
    const int tid = threadIdx.x;
    const int v   = blockIdx.x * 256 + tid;

    __shared__ float sB[Kk * 256];

    const float* bpB = bp + (size_t)b * Kk * V;
    #pragma unroll 4
    for (int k2 = 0; k2 < Kk; ++k2)
        sB[k2 * 256 + tid] = bpB[(size_t)k2 * V + v];
    __syncthreads();

    const float  cv = c[(size_t)b * V + v];
    const float* aB = a + (size_t)b * Kk * V;

    for (int k1 = 0; k1 < Kk; ++k1) {
        const float av = aB[(size_t)k1 * V + v] + cv;
        float s = 0.0f;
        #pragma unroll 6
        for (int k2 = 0; k2 < Kk; ++k2)
            s += fmaxf(av + sB[k2 * 256 + tid], 0.0f);
        const size_t idx = ((size_t)b * Kk + k1) * V + v;
        __bf16 h = (__bf16)s;
        xH[idx] = h;
        xL[idx] = (__bf16)(s - (float)h);
    }
}

// ---------------------------------------------------------------------------
// Host-side launch
// ---------------------------------------------------------------------------
extern "C" void kernel_launch(void* const* d_in, const int* in_sizes, int n_in,
                              void* d_out, int out_size, void* d_ws, size_t ws_size,
                              hipStream_t stream)
{
    const float* v  = (const float*)d_in[0];  // (64, 36, 2048)
    const float* q  = (const float*)d_in[1];  // (64, 1024)
    const float* W1 = (const float*)d_in[2];  // (2048, 5120)
    const float* b1 = (const float*)d_in[3];  // (2048,)
    const float* W2 = (const float*)d_in[4];  // (2048, 2048)
    const float* b2 = (const float*)d_in[5];  // (2048,)
    const float* W3 = (const float*)d_in[6];  // (1024, 1024)
    const float* b3 = (const float*)d_in[7];  // (1024,)
    float* out = (float*)d_out;               // (64, 36, 2048)

    constexpr int B = 64, Kk = 36, V = 2048, Q = 1024;
    constexpr int Mv = B * Kk;             // 2304 rows (multiple of 16)
    constexpr int W1C = 2 * V + Q;         // 5120

    char* ws = (char*)d_ws;
    size_t off = 0;
    auto carve = [&](size_t bytes) -> char* {
        char* p = ws + off;
        off += (bytes + 255) & ~(size_t)255;
        return p;
    };

    __bf16* vH  = (__bf16*)carve((size_t)Mv * V * 2);
    __bf16* vL  = (__bf16*)carve((size_t)Mv * V * 2);
    __bf16* qH  = (__bf16*)carve((size_t)B * Q * 2);
    __bf16* qL  = (__bf16*)carve((size_t)B * Q * 2);
    __bf16* W1H = (__bf16*)carve((size_t)V * W1C * 2);
    __bf16* W1L = (__bf16*)carve((size_t)V * W1C * 2);
    __bf16* W2H = (__bf16*)carve((size_t)V * V * 2);
    __bf16* W2L = (__bf16*)carve((size_t)V * V * 2);
    __bf16* W3H = (__bf16*)carve((size_t)Q * Q * 2);
    __bf16* W3L = (__bf16*)carve((size_t)Q * Q * 2);
    __bf16* qeH = (__bf16*)carve((size_t)B * Q * 2);
    __bf16* qeL = (__bf16*)carve((size_t)B * Q * 2);
    float*  cB  = (float* )carve((size_t)B * V * 4);
    float*  aB  = (float* )carve((size_t)Mv * V * 4);
    float*  bpB = (float* )carve((size_t)Mv * V * 4);
    __bf16* xH  = (__bf16*)carve((size_t)Mv * V * 2);
    __bf16* xL  = (__bf16*)carve((size_t)Mv * V * 2);
    (void)ws_size;

    auto split = [&](const float* src, __bf16* h, __bf16* l, size_t n) {
        dim3 g((unsigned)((n + 255) / 256));
        split_bf16_kernel<<<g, 256, 0, stream>>>(src, h, l, n);
    };
    split(v,  vH,  vL,  (size_t)Mv * V);
    split(q,  qH,  qL,  (size_t)B * Q);
    split(W1, W1H, W1L, (size_t)V * W1C);
    split(W2, W2H, W2L, (size_t)V * V);
    split(W3, W3H, W3L, (size_t)Q * Q);

    // 1) q_emb = relu(q @ W3^T + b3)  -> split bf16   (M=64, N=1024, K=1024)
    {
        dim3 g(Q / 64, (B + 127) / 128);
        gemm_split_bf16<true, true><<<g, 256, 0, stream>>>(
            qH, qL, Q, W3H, W3L, Q, b3,
            nullptr, qeH, qeL, B, Q, Q, Q);
    }
    // 2) c = q_emb @ Wc^T + b1  (f32)                 (M=64, N=2048, K=1024)
    {
        dim3 g(V / 64, (B + 127) / 128);
        gemm_split_bf16<false, false><<<g, 256, 0, stream>>>(
            qeH, qeL, Q, W1H + 2 * V, W1L + 2 * V, W1C, b1,
            cB, nullptr, nullptr, B, V, Q, V);
    }
    // 3) a = v @ Wa^T  (f32)                          (M=2304, N=2048, K=2048)
    {
        dim3 g(V / 64, (Mv + 127) / 128);
        gemm_split_bf16<false, false><<<g, 256, 0, stream>>>(
            vH, vL, V, W1H, W1L, W1C, nullptr,
            aB, nullptr, nullptr, Mv, V, V, V);
    }
    // 4) bpart = v @ Wb^T  (f32)                      (M=2304, N=2048, K=2048)
    {
        dim3 g(V / 64, (Mv + 127) / 128);
        gemm_split_bf16<false, false><<<g, 256, 0, stream>>>(
            vH, vL, V, W1H + V, W1L + V, W1C, nullptr,
            bpB, nullptr, nullptr, Mv, V, V, V);
    }
    // 5) x[b,k1,v] = sum_k2 relu(a + bpart + c)  -> split bf16
    {
        dim3 g(V / 256, B);
        pairsum_kernel<<<g, 256, 0, stream>>>(aB, bpB, cB, xH, xL);
    }
    // 6) out = relu(x @ W2^T + b2)  (f32 -> d_out)    (M=2304, N=2048, K=2048)
    {
        dim3 g(V / 64, (Mv + 127) / 128);
        gemm_split_bf16<true, false><<<g, 256, 0, stream>>>(
            xH, xL, V, W2H, W2L, V, b2,
            out, nullptr, nullptr, Mv, V, V, V);
    }
    (void)in_sizes; (void)n_in; (void)out_size;
}
